// Unify_input_33208687133289
// MI455X (gfx1250) — matrix operand order repaired and checked by reading, not compiled, not run on hardware
//
#include <hip/hip_runtime.h>
#include <hip/hip_bf16.h>

typedef __attribute__((ext_vector_type(16))) _Float16 v16h;
typedef __attribute__((ext_vector_type(8)))  float    v8f;

// ---------------- workspace layout ----------------
// f16 packed-weight region (halves), then f32 region.
#define PH_SE_W1 0
#define PH_SE_W2 32768
#define PH_RE_W1 49152
#define PH_RE_W2 73728
#define PH_RA_W  106496
#define PH_AE_W1 155648
#define PH_AE_W2 221184
#define PACK_BYTES (524288)

#define F_S0   0          // 2048*128
#define F_RAVR 262144     // 2048*128   indexed by (b,i)
#define F_RMAX 524288     // 2048*128
#define F_R0   786432     // 2048*256   r[b, agent_id, j, :]
#define F_S1   1310720    // 2048*128
#define F_AE   1572864    // 2048*256   ae[b, agent_id, j, :]

// ---------------- fragment helpers ----------------
// A-matrix 16x32 f16 fragment (ISA 7.12.2): lane = M + 16*group, group=(k>>3)&1
// half-slot h: kp = k-8*group in {0..7}u{16..23}; p = kp<8 ? kp>>1 : 4+((kp-16)>>1); h=2p+(kp&1)
__device__ __forceinline__ void stage_frag(_Float16* lds, int M, int k, float v) {
    int kb  = k >> 5;
    int k32 = k & 31;
    int g   = (k32 >> 3) & 1;
    int kp  = k32 - 8 * g;
    int p   = (kp < 8) ? (kp >> 1) : (4 + ((kp - 16) >> 1));
    int h   = 2 * p + (kp & 1);
    lds[kb * 512 + (M + 16 * g) * 16 + h] = (_Float16)v;
}

// one 16x16 output tile: A from LDS fragments, B from packed global weights
__device__ __forceinline__ v8f gemm_tile(const _Float16* ldsA, const _Float16* wp,
                                         int nt, int KB, int lane) {
    v8f acc = {};
    for (int kb = 0; kb < KB; ++kb) {
        v16h a = *(const v16h*)(ldsA + kb * 512 + lane * 16);
        v16h b = *(const v16h*)(wp + ((size_t)(nt * KB + kb) * 32 + lane) * 16);
        acc = __builtin_amdgcn_wmma_f32_16x16x32_f16(false, a, false, b,
                                                     (short)0, acc, false, false);
    }
    return acc;
}

__device__ __forceinline__ v8f bias_relu(v8f acc, float bias) {
    #pragma unroll
    for (int r = 0; r < 8; ++r) acc[r] = fmaxf(acc[r] + bias, 0.0f);
    return acc;
}

// scatter a 16x16 f32 C/D tile (lane: N=l%16, rows M=r+8*(l/16)) into A-fragment LDS
__device__ __forceinline__ void stage_acc(_Float16* lds, v8f acc, int nt, int lane) {
    int n = lane & 15, g = lane >> 4;
    #pragma unroll
    for (int r = 0; r < 8; ++r) stage_frag(lds, r + 8 * g, nt * 16 + n, acc[r]);
}

__device__ __forceinline__ void store_rows(float* out, int ldN, int colBase, v8f acc,
                                           int lane, long rowBase) {
    int n = lane & 15, g = lane >> 4;
    #pragma unroll
    for (int r = 0; r < 8; ++r)
        out[(size_t)(rowBase + r + 8 * g) * ldN + colBase + n] = acc[r];
}

// ---------------- kernel 0: pack f32 weight (K x N) into WMMA-B f16 fragments ----
// pack[((nt*KT + kt)*32 + lane)*16 + h] = W[kt*32 + (lane>>4)*8 + (h<8 ? h : 8+h)][nt*16 + (lane&15)]
__global__ void k_pack(const float* __restrict__ W, _Float16* __restrict__ out,
                       int K, int N) {
    int total = K * N;
    int KT = K >> 5;
    for (int idx = blockIdx.x * blockDim.x + threadIdx.x; idx < total;
         idx += gridDim.x * blockDim.x) {
        int h = idx & 15, lane = (idx >> 4) & 31, rest = idx >> 9;
        int kt = rest % KT, nt = rest / KT;
        int k = kt * 32 + (lane >> 4) * 8 + (h < 8 ? h : 8 + h);
        int n = nt * 16 + (lane & 15);
        out[idx] = (_Float16)W[(size_t)k * N + n];
    }
}

// ---------------- kernel 1: s0 = mlp2(states, se_*) ; 2048 rows -----------------
__global__ __launch_bounds__(32) void k_se(const float* __restrict__ states,
                                           const float* __restrict__ b1,
                                           const float* __restrict__ b2,
                                           const _Float16* __restrict__ pack,
                                           float* __restrict__ fws) {
    __shared__ __align__(32) _Float16 ldsA[8 * 512];
    __shared__ __align__(32) _Float16 ldsH[4 * 512];
    int lane = threadIdx.x;
    long tile = blockIdx.x;                 // rows g = tile*16 + M
    for (int idx = lane; idx < 16 * 256; idx += 32) {
        int M = idx >> 8, k = idx & 255;
        stage_frag(ldsA, M, k, states[(size_t)(tile * 16 + M) * 256 + k]);
    }
    __syncthreads();
    for (int nt = 0; nt < 8; ++nt) {
        v8f acc = gemm_tile(ldsA, pack + PH_SE_W1, nt, 8, lane);
        stage_acc(ldsH, bias_relu(acc, b1[nt * 16 + (lane & 15)]), nt, lane);
    }
    __syncthreads();
    for (int nt = 0; nt < 8; ++nt) {
        v8f acc = gemm_tile(ldsH, pack + PH_SE_W2, nt, 4, lane);
        store_rows(fws + F_S0, 128, nt * 16,
                   bias_relu(acc, b2[nt * 16 + (lane & 15)]), lane, tile * 16);
    }
}

// ---------------- kernel 2: relation MLP + masked reductions --------------------
// grid = B*A (one workgroup per (b,i)), 4 waves: wave jt handles rows j=jt*16..+16
__global__ __launch_bounds__(128) void k_rel(const float* __restrict__ relations,
                                             const float* __restrict__ alive,
                                             const float* __restrict__ b1,
                                             const float* __restrict__ b2,
                                             const int* __restrict__ agent_id_p,
                                             const _Float16* __restrict__ pack,
                                             float* __restrict__ fws) {
    __shared__ __align__(32) _Float16 ldsA[4][6 * 512];
    __shared__ __align__(32) _Float16 ldsH[4][4 * 512];
    __shared__ float redAvr[128];
    __shared__ int   redMax[128];
    int tid = threadIdx.x, lane = tid & 31, jt = tid >> 5;
    long bi = blockIdx.x;
    int b = (int)(bi >> 6), i = (int)(bi & 63);
    if (tid < 128) { redAvr[tid] = 0.0f; redMax[tid] = 0; }
    __syncthreads();

    const float* s0 = fws + F_S0;
    _Float16* A = ldsA[jt];
    for (int idx = lane; idx < 16 * 192; idx += 32) {
        int M = idx / 192, k = idx % 192;
        int j = jt * 16 + M;
        float v = (k < 64) ? relations[((size_t)bi * 64 + j) * 64 + k]
                           : s0[(size_t)(b * 64 + j) * 128 + (k - 64)];
        stage_frag(A, M, k, v);
    }
    __syncthreads();

    _Float16* Hh = ldsH[jt];
    for (int nt = 0; nt < 8; ++nt) {
        v8f acc = gemm_tile(A, pack + PH_RE_W1, nt, 6, lane);
        stage_acc(Hh, bias_relu(acc, b1[nt * 16 + (lane & 15)]), nt, lane);
    }
    __syncthreads();

    int g = lane >> 4, n = lane & 15;
    float mk[8];
    #pragma unroll
    for (int r = 0; r < 8; ++r) mk[r] = alive[b * 64 + jt * 16 + r + 8 * g];
    int agent = *agent_id_p;
    float* r0 = fws + F_R0;

    for (int half = 0; half < 2; ++half) {
        for (int nt = 0; nt < 8; ++nt) {
            int NT = half * 8 + nt;
            v8f acc = gemm_tile(Hh, pack + PH_RE_W2, NT, 4, lane);
            acc = bias_relu(acc, b2[NT * 16 + n]);
            int col = NT * 16 + n;
            if (i == agent) {                       // keep unmasked r for ae stage
                #pragma unroll
                for (int r = 0; r < 8; ++r)
                    r0[(size_t)(b * 64 + jt * 16 + r + 8 * g) * 256 + col] = acc[r];
            }
            if (half == 0) {
                float s = 0.0f;
                #pragma unroll
                for (int r = 0; r < 8; ++r) s += acc[r] * mk[r];
                atomicAdd(&redAvr[col], s);
            } else {
                float m = 0.0f;                     // masked values, post-relu >= 0
                #pragma unroll
                for (int r = 0; r < 8; ++r) m = fmaxf(m, acc[r] * mk[r]);
                atomicMax(&redMax[col - 128], __float_as_int(m));
            }
        }
    }
    __syncthreads();
    if (tid < 128) {
        fws[F_RAVR + (size_t)bi * 128 + tid] = redAvr[tid] * (1.0f / 64.0f);
        fws[F_RMAX + (size_t)bi * 128 + tid] = __int_as_float(redMax[tid]);
    }
}

// ---------------- kernel 3: s1 = relu([s0, r_avr, r_max] @ ra_w + ra_b) ---------
__global__ __launch_bounds__(32) void k_ra(const float* __restrict__ rb,
                                           const _Float16* __restrict__ pack,
                                           float* __restrict__ fws) {
    __shared__ __align__(32) _Float16 ldsA[12 * 512];
    int lane = threadIdx.x;
    long tile = blockIdx.x;
    const float* s0 = fws + F_S0;
    const float* ra = fws + F_RAVR;
    const float* rm = fws + F_RMAX;
    for (int idx = lane; idx < 16 * 384; idx += 32) {
        int M = idx / 384, k = idx % 384;
        size_t gr = (size_t)tile * 16 + M;
        float v = (k < 128) ? s0[gr * 128 + k]
                : (k < 256) ? ra[gr * 128 + (k - 128)]
                            : rm[gr * 128 + (k - 256)];
        stage_frag(ldsA, M, k, v);
    }
    __syncthreads();
    for (int nt = 0; nt < 8; ++nt) {
        v8f acc = gemm_tile(ldsA, pack + PH_RA_W, nt, 12, lane);
        store_rows(fws + F_S1, 128, nt * 16,
                   bias_relu(acc, rb[nt * 16 + (lane & 15)]), lane, tile * 16);
    }
}

// ---------------- kernel 4: ae rows for i = agent_id only -----------------------
__global__ __launch_bounds__(32) void k_ae(const float* __restrict__ b1,
                                           const float* __restrict__ b2,
                                           const int* __restrict__ agent_id_p,
                                           const _Float16* __restrict__ pack,
                                           float* __restrict__ fws) {
    __shared__ __align__(32) _Float16 ldsA[16 * 512];
    __shared__ __align__(32) _Float16 ldsH[4 * 512];
    int lane = threadIdx.x;
    int tile = blockIdx.x;                  // b = tile>>2 ; jt = tile&3
    int b = tile >> 2, jt = tile & 3;
    int agent = *agent_id_p;
    const float* r0 = fws + F_R0;
    const float* s1 = fws + F_S1;
    for (int idx = lane; idx < 16 * 512; idx += 32) {
        int M = idx >> 9, k = idx & 511;
        int j = jt * 16 + M;
        float v;
        if (k < 256)      v = r0[(size_t)(b * 64 + j) * 256 + k];
        else if (k < 384) v = s1[(size_t)(b * 64 + j) * 128 + (k - 256)];
        else              v = s1[(size_t)(b * 64 + agent) * 128 + (k - 384)];
        stage_frag(ldsA, M, k, v);
    }
    __syncthreads();
    for (int nt = 0; nt < 8; ++nt) {
        v8f acc = gemm_tile(ldsA, pack + PH_AE_W1, nt, 16, lane);
        stage_acc(ldsH, bias_relu(acc, b1[nt * 16 + (lane & 15)]), nt, lane);
    }
    __syncthreads();
    for (int NT = 0; NT < 16; ++NT) {
        v8f acc = gemm_tile(ldsH, pack + PH_AE_W2, NT, 4, lane);
        store_rows(fws + F_AE, 256, NT * 16,
                   bias_relu(acc, b2[NT * 16 + (lane & 15)]), lane,
                   (long)b * 64 + jt * 16);
    }
}

// ---------------- kernel 5: selection epilogue + outputs ------------------------
__global__ void k_final(const float* __restrict__ aem, const int* __restrict__ action,
                        const int* __restrict__ agent_id_p,
                        const float* __restrict__ fws, float* __restrict__ out) {
    int idx = blockIdx.x * blockDim.x + threadIdx.x;
    if (idx >= 32 * 64 * 128) return;
    int c = idx & 127, j = (idx >> 7) & 63, b = idx >> 13;
    int act = action[b], agent = *agent_id_p;
    const float* s1 = fws + F_S1;
    const float* ae = fws + F_AE;
    float sel = (act < 16) ? aem[(size_t)act * 128 + c]
                           : ae[(size_t)(b * 64 + (act - 16)) * 256 + c];       // active half
    float pas = (act == 16 + j) ? ae[(size_t)(b * 64 + j) * 256 + 128 + c] : 0.0f;
    float st  = s1[(size_t)(b * 64 + j) * 128 + c] + ((j == agent) ? sel : 0.0f) + pas;
    out[idx] = st;                                  // state
    out[32 * 64 * 128 + 4096 + idx] = pas;          // passive_sel
    if (j == 0) out[32 * 64 * 128 + b * 128 + c] = sel;   // sel (B,1,1,H)
}

extern "C" void kernel_launch(void* const* d_in, const int* in_sizes, int n_in,
                              void* d_out, int out_size, void* d_ws, size_t ws_size,
                              hipStream_t stream) {
    const float* states  = (const float*)d_in[0];
    const float* rels    = (const float*)d_in[1];
    const float* alive   = (const float*)d_in[2];
    const float* se_w1   = (const float*)d_in[3];
    const float* se_b1   = (const float*)d_in[4];
    const float* se_w2   = (const float*)d_in[5];
    const float* se_b2   = (const float*)d_in[6];
    const float* re_w1   = (const float*)d_in[7];
    const float* re_b1   = (const float*)d_in[8];
    const float* re_w2   = (const float*)d_in[9];
    const float* re_b2   = (const float*)d_in[10];
    const float* ra_w    = (const float*)d_in[11];
    const float* ra_b    = (const float*)d_in[12];
    const float* ae_w1   = (const float*)d_in[13];
    const float* ae_b1   = (const float*)d_in[14];
    const float* ae_w2   = (const float*)d_in[15];
    const float* ae_b2   = (const float*)d_in[16];
    const float* aem     = (const float*)d_in[17];
    const int*   action  = (const int*)d_in[19];
    const int*   agentid = (const int*)d_in[20];

    _Float16* pack = (_Float16*)d_ws;
    float*    fws  = (float*)((char*)d_ws + PACK_BYTES);

    k_pack<<<64, 256, 0, stream>>>(se_w1, pack + PH_SE_W1, 256, 128);
    k_pack<<<64, 256, 0, stream>>>(se_w2, pack + PH_SE_W2, 128, 128);
    k_pack<<<64, 256, 0, stream>>>(re_w1, pack + PH_RE_W1, 192, 128);
    k_pack<<<64, 256, 0, stream>>>(re_w2, pack + PH_RE_W2, 128, 256);
    k_pack<<<64, 256, 0, stream>>>(ra_w,  pack + PH_RA_W,  384, 128);
    k_pack<<<64, 256, 0, stream>>>(ae_w1, pack + PH_AE_W1, 512, 128);
    k_pack<<<64, 256, 0, stream>>>(ae_w2, pack + PH_AE_W2, 128, 256);

    k_se <<<128, 32, 0, stream>>>(states, se_b1, se_b2, pack, fws);
    k_rel<<<32 * 64, 128, 0, stream>>>(rels, alive, re_b1, re_b2, agentid, pack, fws);
    k_ra <<<128, 32, 0, stream>>>(ra_b, pack, fws);
    k_ae <<<128, 32, 0, stream>>>(ae_b1, ae_b2, agentid, pack, fws);
    k_final<<<(32 * 64 * 128 + 255) / 256, 256, 0, stream>>>(aem, action, agentid,
                                                             fws, (float*)d_out);
}
